// DAT_58652073394664
// MI455X (gfx1250) — matrix operand order repaired and checked by reading, not compiled, not run on hardware
//
#include <hip/hip_runtime.h>

// ---------------------------------------------------------------------------
// CDNA5 (gfx1250) bf16 WMMA implementation of the EDA/CSWin transformer block.
// All matrix math goes through v_wmma_f32_16x16x32_bf16 (fp32 accumulate).
// Contiguous tile staging uses GLOBAL_LOAD_ASYNC_TO_LDS_B128 (ASYNCcnt-tracked)
// via __builtin_amdgcn_global_load_async_to_lds_b128 (guarded; falls back to
// load + ds_store otherwise).
// ---------------------------------------------------------------------------

typedef __bf16 bf16_t;
typedef __attribute__((ext_vector_type(16))) __bf16 v16bf;
typedef __attribute__((ext_vector_type(8)))  __bf16 bf16x8;
typedef __attribute__((ext_vector_type(8)))  float  v8f;

#define DEVINL __device__ __forceinline__

#ifndef __has_builtin
#define __has_builtin(x) 0
#endif
#if __has_builtin(__builtin_amdgcn_global_load_async_to_lds_b128)
#define USE_ASYNC_LDS 1
#else
#define USE_ASYNC_LDS 0
#endif
#if __has_builtin(__builtin_amdgcn_s_wait_asynccnt)
#define HAVE_WAIT_ASYNC 1
#else
#define HAVE_WAIT_ASYNC 0
#endif

// b128 async copies take pointers to 16-byte int vectors in AS1 (global) and
// AS3 (LDS) — exact types per the clang builtin signature.
typedef int v4i_vs __attribute__((vector_size(16)));
typedef __attribute__((address_space(1))) v4i_vs as1_v4i;
typedef __attribute__((address_space(3))) v4i_vs as3_v4i;

static constexpr int BB   = 8;
static constexpr int CC   = 256;
static constexpr int HH   = 56;
static constexpr int LTOK = HH * HH;        // 3136
static constexpr int NT   = BB * LTOK;      // 25088 tokens (multiple of 128)
static constexpr int ATTN = 392;            // window token count (56*7)
static constexpr float ATT_SCALE = 0.17677669529663689f; // 1/sqrt(32)

DEVINL bf16_t f2bf(float f) {
  union { float f; unsigned int u; } in{f};
  unsigned int u = in.u + 0x7FFFu + ((in.u >> 16) & 1u);  // RNE
  union { unsigned short s; bf16_t b; } out;
  out.s = (unsigned short)(u >> 16);
  return out.b;
}
DEVINL float bf2f(bf16_t b) {
  union { unsigned short s; bf16_t b; } in; in.b = b;
  union { unsigned int u; float f; } out; out.u = ((unsigned int)in.s) << 16;
  return out.f;
}

// Async 16B copy global -> LDS (ASYNCcnt-tracked). No-op stub if unavailable.
DEVINL void async_copy_b128(void* lds_dst, const void* gsrc) {
#if USE_ASYNC_LDS
  __builtin_amdgcn_global_load_async_to_lds_b128((as1_v4i*)gsrc,
                                                 (as3_v4i*)lds_dst, 0, 0);
#else
  (void)lds_dst; (void)gsrc;
#endif
}
DEVINL void wait_async0() {
#if USE_ASYNC_LDS
#if HAVE_WAIT_ASYNC
  __builtin_amdgcn_s_wait_asynccnt(0);
#else
  asm volatile("s_wait_asynccnt 0x0" ::: "memory");
#endif
#endif
}

union Frag16 { bf16x8 h[2]; v16bf v; };

// A-matrix fragment (16x32 bf16). Per ISA layout: lane -> M row (lane&15),
// kbase = (lane&16)?8:0; element e -> K = kbase + e (+8 for e>=8).
// With row-major [row][k] LDS storage this is two contiguous 8-element loads.
DEVINL v16bf load_a_frag(const bf16_t* base, int stride, int row, int lane) {
  const int kb = (lane & 16) ? 8 : 0;
  Frag16 f;
  f.h[0] = *(const bf16x8*)(base + row * stride + kb);
  f.h[1] = *(const bf16x8*)(base + row * stride + kb + 16);
  return f.v;
}
// B-matrix fragment (32x16 bf16) read from a [n][k] (transposed) LDS buffer:
// lane -> N col (lane&15), kbase = (lane&16)?8:0; element e -> K = kbase + e (+8 for e>=8).
DEVINL v16bf load_bt_frag(const bf16_t* base, int stride, int ncol, int lane) {
  const int kb = (lane & 16) ? 8 : 0;
  const int n  = lane & 15;
  Frag16 f;
  f.h[0] = *(const bf16x8*)(base + (ncol + n) * stride + kb);
  f.h[1] = *(const bf16x8*)(base + (ncol + n) * stride + kb + 16);
  return f.v;
}

DEVINL v8f wmma_bf16(v16bf a, v16bf b, v8f c) {
  return __builtin_amdgcn_wmma_f32_16x16x32_bf16(false, a, false, b, (short)0, c,
                                                 false, false);
}

// ---------------------------------------------------------------------------
// Kernel 1: NCHW -> (B,L,C) fp32 transpose (coalesced via 32x32 LDS tile).
// ---------------------------------------------------------------------------
__global__ __launch_bounds__(256) void k_nchw2nlc(const float* __restrict__ x,
                                                  float* __restrict__ xtok) {
  __shared__ float tile[32][33];
  const int l0 = blockIdx.x * 32, c0 = blockIdx.y * 32, b = blockIdx.z;
  const int tx = threadIdx.x, ty = threadIdx.y;
#pragma unroll
  for (int j = 0; j < 4; ++j) {
    int c = c0 + ty + j * 8;
    tile[ty + j * 8][tx] = x[(size_t)(b * CC + c) * LTOK + l0 + tx];
  }
  __syncthreads();
#pragma unroll
  for (int j = 0; j < 4; ++j) {
    int l = l0 + ty + j * 8;
    xtok[((size_t)b * LTOK + l) * CC + c0 + tx] = tile[tx][ty + j * 8];
  }
}

// ---------------------------------------------------------------------------
// Kernel 2: LayerNorm (one wave32 per token, C=256 -> 8 elems/lane) -> bf16.
// ---------------------------------------------------------------------------
__global__ __launch_bounds__(256) void k_ln(const float* __restrict__ src,
                                            const float* __restrict__ g,
                                            const float* __restrict__ bsh,
                                            bf16_t* __restrict__ dst) {
  const int lane = threadIdx.x & 31, w = threadIdx.x >> 5;
  const size_t tok = (size_t)blockIdx.x * 8 + w;
  const float* row = src + tok * CC;
  float v[8], s = 0.f, s2 = 0.f;
#pragma unroll
  for (int i = 0; i < 8; ++i) {
    v[i] = row[lane + 32 * i];
    s += v[i];
    s2 += v[i] * v[i];
  }
#pragma unroll
  for (int off = 1; off < 32; off <<= 1) {
    s  += __shfl_xor(s, off, 32);
    s2 += __shfl_xor(s2, off, 32);
  }
  const float mu = s * (1.0f / CC);
  const float var = s2 * (1.0f / CC) - mu * mu;
  const float rs = rsqrtf(var + 1e-5f);
#pragma unroll
  for (int i = 0; i < 8; ++i) {
    int c = lane + 32 * i;
    dst[tok * CC + c] = f2bf((v[i] - mu) * rs * g[c] + bsh[c]);
  }
}

// ---------------------------------------------------------------------------
// Kernel 3: fp32 -> bf16 weight conversion.
// ---------------------------------------------------------------------------
__global__ void k_cvt(const float* __restrict__ s, bf16_t* __restrict__ d, int n) {
  int i = blockIdx.x * blockDim.x + threadIdx.x;
  if (i < n) d[i] = f2bf(s[i]);
}

// ---------------------------------------------------------------------------
// Kernel 4: bf16 WMMA GEMM, BM=128 BN=64 BK=32, 8 waves (4x2), 2x2 frags/wave.
// A tile staged with async global->LDS b128; B tile staged transposed.
// Epilogues: 0 = bf16 out; 1 = +bias+res -> f32; 2 = +bias, GELU -> bf16;
//            3 = +bias+res -> f32 NCHW (final output).
// ---------------------------------------------------------------------------
template <int EPI>
__global__ __launch_bounds__(256) void k_gemm(const bf16_t* __restrict__ A,
                                              const bf16_t* __restrict__ Bw,
                                              const float* __restrict__ bias,
                                              const float* __restrict__ res,
                                              void* __restrict__ dstv,
                                              int M, int N, int K) {
  __shared__ bf16_t As[128][40];  // [m][k]
  __shared__ bf16_t Bt[64][40];   // [n][k] (transposed)
  const int tid = threadIdx.x, lane = tid & 31, w = tid >> 5;
  const int wm = w & 3, wn = w >> 2;  // 4x2 wave grid
  const int mBase = blockIdx.y * 128, nBase = blockIdx.x * 64;
  v8f acc[2][2] = {};

  for (int kt = 0; kt < K; kt += 32) {
#pragma unroll
    for (int i = 0; i < 2; ++i) {  // stage A (128x32), 16B per lane per chunk
      int q = tid + i * 256;
      int r = q >> 2, cc = (q & 3) * 8;
      const bf16_t* src = &A[(size_t)(mBase + r) * K + kt + cc];
#if USE_ASYNC_LDS
      async_copy_b128(&As[r][cc], src);
#else
      *(bf16x8*)&As[r][cc] = *(const bf16x8*)src;
#endif
    }
    {  // stage B transposed (32x64 -> [n][k])
      int kk = tid >> 3, nc = (tid & 7) * 8;
      bf16x8 v = *(const bf16x8*)&Bw[(size_t)(kt + kk) * N + nBase + nc];
#pragma unroll
      for (int i = 0; i < 8; ++i) Bt[nc + i][kk] = v[i];
    }
    if (kt + 32 < K) {  // prefetch next k-tile (global_prefetch_b8)
      __builtin_prefetch(&A[(size_t)(mBase + (tid >> 1)) * K + kt + 32 + (tid & 1) * 16], 0, 0);
      __builtin_prefetch(&Bw[(size_t)(kt + 32 + (tid >> 3)) * N + nBase + (tid & 7) * 8], 0, 0);
    }
    wait_async0();
    __syncthreads();
    v16bf af[2], bf[2];
#pragma unroll
    for (int mi = 0; mi < 2; ++mi)
      af[mi] = load_a_frag(&As[0][0], 40, wm * 32 + mi * 16 + (lane & 15), lane);
#pragma unroll
    for (int ni = 0; ni < 2; ++ni)
      bf[ni] = load_bt_frag(&Bt[0][0], 40, wn * 32 + ni * 16, lane);
#pragma unroll
    for (int mi = 0; mi < 2; ++mi)
#pragma unroll
      for (int ni = 0; ni < 2; ++ni)
        acc[mi][ni] = wmma_bf16(af[mi], bf[ni], acc[mi][ni]);
    __syncthreads();
  }

  // Epilogue. C/D layout: element v -> row = v + ((lane&16)?8:0), col = lane&15.
  const int hi8 = (lane & 16) ? 8 : 0;
  const int nl = lane & 15;
#pragma unroll
  for (int mi = 0; mi < 2; ++mi)
#pragma unroll
    for (int ni = 0; ni < 2; ++ni) {
      const int r0 = mBase + wm * 32 + mi * 16 + hi8;
      const int n = nBase + wn * 32 + ni * 16 + nl;
#pragma unroll
      for (int v = 0; v < 8; ++v) {
        const int m = r0 + v;
        float a = acc[mi][ni][v];
        if (EPI == 0) {
          ((bf16_t*)dstv)[(size_t)m * N + n] = f2bf(a);
        } else if (EPI == 1) {
          a += bias[n] + res[(size_t)m * N + n];
          ((float*)dstv)[(size_t)m * N + n] = a;
        } else if (EPI == 2) {
          a += bias[n];
          a = 0.5f * a * (1.0f + erff(a * 0.70710678118654752f));  // exact GELU
          ((bf16_t*)dstv)[(size_t)m * N + n] = f2bf(a);
        } else {
          a += bias[n] + res[(size_t)m * CC + n];
          const int b = m / LTOK, l = m - b * LTOK;
          ((float*)dstv)[((size_t)(b * CC + n)) * LTOK + l] = a;  // NCHW
        }
      }
    }
}

// ---------------------------------------------------------------------------
// Kernel 5: depthwise 3x3 LePE conv on v (bf16 in qkv, channels 512..639).
// Output fp32 (B,L,256) with channel = branch*128 + ch.
// ---------------------------------------------------------------------------
__global__ __launch_bounds__(256) void k_lepe(const bf16_t* __restrict__ qkv,
                                              const float* __restrict__ w0,
                                              const float* __restrict__ b0,
                                              const float* __restrict__ w1,
                                              const float* __restrict__ b1,
                                              float* __restrict__ lepe) {
  const int pix = blockIdx.x;           // b*3136 + l
  const int b = pix / LTOK, l = pix - b * LTOK;
  const int y = l / HH, x = l - y * HH;
  const int t = threadIdx.x;
  const int branch = t >> 7, ch = t & 127;
  const float* wgt = branch ? w1 : w0;
  const float* bia = branch ? b1 : b0;
  float acc = bia[ch];
#pragma unroll
  for (int dy = -1; dy <= 1; ++dy)
#pragma unroll
    for (int dx = -1; dx <= 1; ++dx) {
      int yy = y + dy, xx = x + dx;
      if (yy >= 0 && yy < HH && xx >= 0 && xx < HH) {
        float vv = bf2f(qkv[((size_t)b * LTOK + yy * HH + xx) * 640 + 512 + ch]);
        acc += vv * wgt[((dy + 1) * 3 + (dx + 1)) * 128 + ch];
      }
    }
  lepe[(size_t)pix * CC + t] = acc;
}

// ---------------------------------------------------------------------------
// Kernel 6: windowed flash attention (both branches), WMMA bf16.
// grid = (7 m-tiles, 64 = B*win, 8 = branch*4 + head), 128 threads (4 waves).
// Each wave owns a 16-row strip of the 64-row Q tile; KV streamed in 32-key
// tiles (392 padded to 416 with -1e30 masking), online softmax.
// ---------------------------------------------------------------------------
__global__ __launch_bounds__(128) void k_attn(const bf16_t* __restrict__ qkv,
                                              const float* __restrict__ lepe,
                                              bf16_t* __restrict__ attn_cat) {
  __shared__ bf16_t Qs[64][40];       // [m][d]
  __shared__ bf16_t Ks[32][40];       // [key][d]
  __shared__ bf16_t Vts[32][40];      // [d][key] (transposed)
  __shared__ bf16_t Ps[4][16][40];    // per-wave P staging [m][j]

  const int tid = threadIdx.x, lane = tid & 31, w = tid >> 5;
  const int mtile = blockIdx.x;
  const int bidx = blockIdx.y >> 3, win = blockIdx.y & 7;
  const int branch = blockIdx.z >> 2, head = blockIdx.z & 3;
  const int H_sp = branch ? 7 : HH;
  const int W_sp = branch ? HH : 7;
  const int nWx = HH / W_sp;                 // branch0: 8, branch1: 1
  const int wy = win / nWx, wx = win - wy * nWx;
  const int qoff = branch * 256 + head * 32;
  const int koff = qoff + 128;
  const int voff = 512 + head * 32;
  const int hi8 = (lane & 16) ? 8 : 0;
  const int nl = lane & 15;
  const bf16x8 z8 = {};

  // Stage Q tile (64x32), rows = window tokens mtile*64 ..
#pragma unroll
  for (int i = 0; i < 2; ++i) {
    int q = tid + i * 128;
    int r = q >> 2, cc = (q & 3) * 8;
    int tok = mtile * 64 + r;
    if (tok < ATTN) {
      int rr = tok / W_sp, c2 = tok - rr * W_sp;
      size_t l = (size_t)(wy * H_sp + rr) * HH + wx * W_sp + c2;
      const bf16_t* src = qkv + ((size_t)bidx * LTOK + l) * 640 + qoff + cc;
#if USE_ASYNC_LDS
      async_copy_b128(&Qs[r][cc], src);
#else
      *(bf16x8*)&Qs[r][cc] = *(const bf16x8*)src;
#endif
    } else {
      *(bf16x8*)&Qs[r][cc] = z8;
    }
  }
  wait_async0();
  __syncthreads();
  // Fold softmax scale into Q once (bf16 elementwise, 16 muls total).
  v16bf a_q = load_a_frag(&Qs[0][0], 40, w * 16 + nl, lane);
#pragma unroll
  for (int i = 0; i < 16; ++i) a_q[i] = f2bf(bf2f(a_q[i]) * ATT_SCALE);

  float mrun[8], lrun[8];
  v8f o0 = {}, o1 = {};
#pragma unroll
  for (int v = 0; v < 8; ++v) { mrun[v] = -1e30f; lrun[v] = 0.f; }

  for (int j0 = 0; j0 < 416; j0 += 32) {
    // Stage K (32x32, async) and V transposed (32x32, manual)
    {
      int r = tid >> 2, cc = (tid & 3) * 8;
      int jj = j0 + r;
      if (jj < ATTN) {
        int rr = jj / W_sp, c2 = jj - rr * W_sp;
        size_t l = (size_t)(wy * H_sp + rr) * HH + wx * W_sp + c2;
        const bf16_t* p = qkv + ((size_t)bidx * LTOK + l) * 640;
#if USE_ASYNC_LDS
        async_copy_b128(&Ks[r][cc], p + koff + cc);
#else
        *(bf16x8*)&Ks[r][cc] = *(const bf16x8*)(p + koff + cc);
#endif
        bf16x8 vv = *(const bf16x8*)(p + voff + cc);
#pragma unroll
        for (int i = 0; i < 8; ++i) Vts[cc + i][r] = vv[i];
      } else {
        *(bf16x8*)&Ks[r][cc] = z8;
#pragma unroll
        for (int i = 0; i < 8; ++i) Vts[cc + i][r] = z8[i];
      }
    }
    wait_async0();
    __syncthreads();

    // S = Q K^T for two 16-wide key sub-tiles (scale pre-folded into Q)
    v8f zz = {};
    v16bf bk0 = load_bt_frag(&Ks[0][0], 40, 0, lane);
    v16bf bk1 = load_bt_frag(&Ks[0][0], 40, 16, lane);
    v8f s0 = wmma_bf16(a_q, bk0, zz);
    v8f s1 = wmma_bf16(a_q, bk1, zz);
    const bool bad0 = (j0 + nl) >= ATTN;
    const bool bad1 = (j0 + 16 + nl) >= ATTN;

    // Online softmax per row (rows live across 16-lane groups in C layout)
#pragma unroll
    for (int v = 0; v < 8; ++v) {
      float e0 = bad0 ? -1e30f : s0[v];
      float e1 = bad1 ? -1e30f : s1[v];
      float rm = fmaxf(e0, e1);
#pragma unroll
      for (int off = 1; off < 16; off <<= 1) rm = fmaxf(rm, __shfl_xor(rm, off, 16));
      float mnew = fmaxf(mrun[v], rm);
      float alpha = __expf(mrun[v] - mnew);
      mrun[v] = mnew;
      float p0 = __expf(e0 - mnew), p1 = __expf(e1 - mnew);
      float rs = p0 + p1;
#pragma unroll
      for (int off = 1; off < 16; off <<= 1) rs += __shfl_xor(rs, off, 16);
      lrun[v] = lrun[v] * alpha + rs;
      o0[v] *= alpha;
      o1[v] *= alpha;
      Ps[w][v + hi8][nl] = f2bf(p0);        // relayout C->A through LDS
      Ps[w][v + hi8][16 + nl] = f2bf(p1);
    }

    // O += P V
    v16bf ap = load_a_frag(&Ps[w][0][0], 40, nl, lane);
    v16bf bv0 = load_bt_frag(&Vts[0][0], 40, 0, lane);
    v16bf bv1 = load_bt_frag(&Vts[0][0], 40, 16, lane);
    o0 = wmma_bf16(ap, bv0, o0);
    o1 = wmma_bf16(ap, bv1, o1);
    __syncthreads();
  }

  // Write out: attn + LePE, bf16, channel = branch*128 + head*32 + d
#pragma unroll
  for (int v = 0; v < 8; ++v) {
    int tok = mtile * 64 + w * 16 + v + hi8;
    if (tok < ATTN) {
      float inv = 1.0f / lrun[v];
      int rr = tok / W_sp, c2 = tok - rr * W_sp;
      size_t l = (size_t)(wy * H_sp + rr) * HH + wx * W_sp + c2;
      size_t base = ((size_t)bidx * LTOK + l) * CC + branch * 128 + head * 32;
      attn_cat[base + nl] = f2bf(o0[v] * inv + lepe[base + nl]);
      attn_cat[base + 16 + nl] = f2bf(o1[v] * inv + lepe[base + 16 + nl]);
    }
  }
}

// ---------------------------------------------------------------------------
// Host-side orchestration
// ---------------------------------------------------------------------------
extern "C" void kernel_launch(void* const* d_in, const int* in_sizes, int n_in,
                              void* d_out, int out_size, void* d_ws, size_t ws_size,
                              hipStream_t stream) {
  (void)in_sizes; (void)n_in; (void)out_size; (void)ws_size;
  const float* x       = (const float*)d_in[0];
  const float* w_qkv   = (const float*)d_in[1];
  const float* g1      = (const float*)d_in[2];
  const float* b1      = (const float*)d_in[3];
  const float* conv_w0 = (const float*)d_in[4];
  const float* conv_b0 = (const float*)d_in[5];
  const float* conv_w1 = (const float*)d_in[6];
  const float* conv_b1 = (const float*)d_in[7];
  const float* w_proj  = (const float*)d_in[8];
  const float* b_proj  = (const float*)d_in[9];
  const float* g2      = (const float*)d_in[10];
  const float* b2      = (const float*)d_in[11];
  const float* w_fc1   = (const float*)d_in[12];
  const float* b_fc1   = (const float*)d_in[13];
  const float* w_fc2   = (const float*)d_in[14];
  const float* b_fc2   = (const float*)d_in[15];

  char* ws = (char*)d_ws;
  const size_t o_xtok = 0;                                   // f32 NT*256
  const size_t o_ln   = o_xtok + (size_t)NT * CC * 4;        // bf16 NT*256 (ln1, later ln2)
  const size_t o_big  = o_ln + (size_t)NT * CC * 2;          // bf16: qkv NT*640, later h1 NT*1024
  const size_t o_lepe = o_big + (size_t)NT * 1024 * 2;       // f32 NT*256
  const size_t o_attn = o_lepe + (size_t)NT * CC * 4;        // bf16 NT*256
  const size_t o_py   = o_attn + (size_t)NT * CC * 2;        // f32 NT*256
  const size_t o_wq   = o_py + (size_t)NT * CC * 4;
  const size_t o_wp   = o_wq + (size_t)256 * 640 * 2;
  const size_t o_wf1  = o_wp + (size_t)256 * 256 * 2;
  const size_t o_wf2  = o_wf1 + (size_t)256 * 1024 * 2;

  float*  xtok  = (float*)(ws + o_xtok);
  bf16_t* lnb   = (bf16_t*)(ws + o_ln);
  bf16_t* qkv   = (bf16_t*)(ws + o_big);
  bf16_t* h1    = (bf16_t*)(ws + o_big);   // aliases qkv (dead by then)
  float*  lepe  = (float*)(ws + o_lepe);
  bf16_t* attnc = (bf16_t*)(ws + o_attn);
  float*  projy = (float*)(ws + o_py);
  bf16_t* wq    = (bf16_t*)(ws + o_wq);
  bf16_t* wp    = (bf16_t*)(ws + o_wp);
  bf16_t* wf1   = (bf16_t*)(ws + o_wf1);
  bf16_t* wf2   = (bf16_t*)(ws + o_wf2);

  // Weight conversion fp32 -> bf16
  k_cvt<<<(256 * 640 + 255) / 256, 256, 0, stream>>>(w_qkv, wq, 256 * 640);
  k_cvt<<<(256 * 256 + 255) / 256, 256, 0, stream>>>(w_proj, wp, 256 * 256);
  k_cvt<<<(256 * 1024 + 255) / 256, 256, 0, stream>>>(w_fc1, wf1, 256 * 1024);
  k_cvt<<<(1024 * 256 + 255) / 256, 256, 0, stream>>>(w_fc2, wf2, 1024 * 256);

  // NCHW -> (B,L,C) fp32 residual
  k_nchw2nlc<<<dim3(LTOK / 32, CC / 32, BB), dim3(32, 8), 0, stream>>>(x, xtok);
  // LN1 -> bf16
  k_ln<<<NT / 8, 256, 0, stream>>>(xtok, g1, b1, lnb);
  // QKV GEMM: (25088x256) x (256x640) -> bf16
  k_gemm<0><<<dim3(640 / 64, NT / 128), 256, 0, stream>>>(
      lnb, wq, nullptr, nullptr, qkv, NT, 640, 256);
  // LePE depthwise conv on v
  k_lepe<<<NT, 256, 0, stream>>>(qkv, conv_w0, conv_b0, conv_w1, conv_b1, lepe);
  // Windowed attention (both branches fused in grid.z)
  k_attn<<<dim3(7, 64, 8), 128, 0, stream>>>(qkv, lepe, attnc);
  // Proj GEMM + bias + residual -> fp32
  k_gemm<1><<<dim3(CC / 64, NT / 128), 256, 0, stream>>>(
      attnc, wp, b_proj, xtok, projy, NT, CC, 256);
  // LN2 -> bf16 (reuses ln buffer)
  k_ln<<<NT / 8, 256, 0, stream>>>(projy, g2, b2, lnb);
  // FC1 + bias + GELU -> bf16 (h1 aliases qkv region)
  k_gemm<2><<<dim3(1024 / 64, NT / 128), 256, 0, stream>>>(
      lnb, wf1, b_fc1, nullptr, h1, NT, 1024, 256);
  // FC2 + bias + residual -> fp32 NCHW final output
  k_gemm<3><<<dim3(CC / 64, NT / 128), 256, 0, stream>>>(
      h1, wf2, b_fc2, projy, d_out, NT, CC, 1024);
}
